// MOEMlp_84920093376644
// MI455X (gfx1250) — matrix operand order repaired and checked by reading, not compiled, not run on hardware
//
#include <hip/hip_runtime.h>
#include <hip/hip_bf16.h>
#include <math.h>

// ---------------- problem constants ----------------
#define E_  23
#define K_  3
#define D_  768
#define H_  3072
#define N_  2048           // 2*1024 tokens
#define NEXP (E_ + 1)      // +1 = shared expert handled uniformly
#define MT_ 32             // tokens per block tile (2 WMMA M-tiles)

typedef __bf16 bf16;
typedef bf16  v16bf __attribute__((ext_vector_type(16)));
typedef bf16  v8bf  __attribute__((ext_vector_type(8)));
typedef float v8f   __attribute__((ext_vector_type(8)));

#define WMMA_BF16(a, b, c) \
    __builtin_amdgcn_wmma_f32_16x16x32_bf16(false, (a), false, (b), (short)0, (c), false, false)

// ---------------- LDS layout for fused expert kernel ----------------
struct SmemExpert {
    int   toks[MT_];
    float wgt[MT_];
    bf16  Bt[2][128][40];      // double-buffered weight tile, TRANSPOSED [col][k] (+pad)
    bf16  Xs[MT_][D_ + 8];     // X tile, row-major bf16
    bf16  Hs[MT_][H_ + 8];     // gelu(x@w1+b1) activation tile, bf16 (on-chip between GEMMs)
};

// packed bf16 pair store (lowers to v_cvt_pk_bf16_f32 + ds_store_b32)
__device__ __forceinline__ void stage_pair(bf16* dst, float lo, float hi) {
    union { bf16 h[2]; unsigned u; } pk;
    pk.h[0] = (bf16)lo; pk.h[1] = (bf16)hi;
    *reinterpret_cast<unsigned*>(dst) = pk.u;
}

// branch-free tanh-form GELU: x * sigmoid(1.5957691x + 0.0713548x^3)
// (v_exp_f32 + v_rcp_f32, ~6 VALU ops; replaces the huge branchy erff expansion)
__device__ __forceinline__ float gelu_fast(float v) {
    float z = v * (1.5957691216057308f + 0.0713548162726009f * v * v);
    return v * __builtin_amdgcn_rcpf(1.f + __expf(-z));
}

// A fragment: K groups at p[0..7] and p[16..23]  (2x ds_load_b128)
__device__ __forceinline__ v16bf ldfragA(const bf16* p) {
    v8bf lo = *reinterpret_cast<const v8bf*>(p);
    v8bf hi = *reinterpret_cast<const v8bf*>(p + 16);
    return __builtin_shufflevector(lo, hi, 0,1,2,3,4,5,6,7,8,9,10,11,12,13,14,15);
}
// B fragment: K groups at p[0..7] and p[16..23]  (2x ds_load_b128)
__device__ __forceinline__ v16bf ldfragB(const bf16* p) {
    v8bf lo = *reinterpret_cast<const v8bf*>(p);
    v8bf hi = *reinterpret_cast<const v8bf*>(p + 16);
    return __builtin_shufflevector(lo, hi, 0,1,2,3,4,5,6,7,8,9,10,11,12,13,14,15);
}

// ---------------- init kernels ----------------
__global__ void zero_out_kernel(float* __restrict__ out) {
    size_t i = (size_t)blockIdx.x * blockDim.x + threadIdx.x;
    if (i < (size_t)N_ * D_) out[i] = 0.f;
}

__global__ void init_small_kernel(int* __restrict__ counts, float* __restrict__ gsum) {
    int t = threadIdx.x;
    if (t < E_) { counts[t] = 0; gsum[t] = 0.f; }
}

// ---------------- gate + top-k dispatch ----------------
__global__ __launch_bounds__(256) void gate_kernel(
    const float* __restrict__ x, const float* __restrict__ gw,
    const float* __restrict__ gb, const float* __restrict__ biasb,
    float* __restrict__ comb, int* __restrict__ counts,
    float* __restrict__ gsum, int* __restrict__ tlist)
{
    const int wave = threadIdx.x >> 5;
    const int lane = threadIdx.x & 31;
    const int n = blockIdx.x * 8 + wave;
    if (n >= N_) return;

    __shared__ float g[8][E_];

    float xreg[D_ / 32];
    const float* xr = x + (size_t)n * D_;
    #pragma unroll
    for (int i = 0; i < D_ / 32; ++i) xreg[i] = xr[lane + i * 32];

    for (int e = 0; e < E_; ++e) {
        float acc = 0.f;
        #pragma unroll
        for (int i = 0; i < D_ / 32; ++i)
            acc += xreg[i] * gw[(size_t)(lane + i * 32) * E_ + e];
        #pragma unroll
        for (int off = 16; off > 0; off >>= 1)
            acc += __shfl_xor(acc, off, 32);
        if (lane == 0)
            g[wave][e] = 1.f / (1.f + expf(-(acc + gb[e])));
    }

    if (lane == 0) {
        float gl[E_]; float sumg = 0.f;
        for (int e = 0; e < E_; ++e) { gl[e] = g[wave][e]; sumg += gl[e]; }

        int idx[K_]; float wv[K_]; bool used[E_];
        for (int e = 0; e < E_; ++e) used[e] = false;
        float wsum = 0.f;
        for (int k = 0; k < K_; ++k) {
            int best = 0; float bv = -1e30f;
            for (int e = 0; e < E_; ++e) {
                if (used[e]) continue;
                float sc = gl[e] + biasb[e];
                if (sc > bv) { bv = sc; best = e; }
            }
            used[best] = true; idx[k] = best; wv[k] = gl[best]; wsum += gl[best];
        }
        for (int e = 0; e < E_; ++e) comb[(size_t)n * E_ + e] = 0.f;
        for (int k = 0; k < K_; ++k) {
            comb[(size_t)n * E_ + idx[k]] = wv[k] / wsum;
            int slot = atomicAdd(&counts[idx[k]], 1);
            tlist[(size_t)idx[k] * N_ + slot] = n;
        }
        float inv = 1.f / sumg;
        for (int e = 0; e < E_; ++e) atomicAdd(&gsum[e], gl[e] * inv);
    }
}

// ---------------- fused expert MLP ----------------
// grid: (N_/MT_ m-tiles, 24). blockIdx.y == E_ is the shared expert.
__global__ __launch_bounds__(256) void expert_kernel(
    const float* __restrict__ x,
    const float* __restrict__ w1, const float* __restrict__ b1,
    const float* __restrict__ w2, const float* __restrict__ b2,
    const float* __restrict__ ws1, const float* __restrict__ bs1,
    const float* __restrict__ ws2, const float* __restrict__ bs2,
    const int* __restrict__ counts, const int* __restrict__ tlist,
    const float* __restrict__ comb, float* __restrict__ out)
{
    extern __shared__ char smem_raw[];
    SmemExpert& s = *reinterpret_cast<SmemExpert*>(smem_raw);

    const int e  = blockIdx.y;
    const int mt = blockIdx.x;
    const bool shared_e = (e == E_);
    const int cnt = shared_e ? N_ : counts[e];
    if (mt * MT_ >= cnt) return;

    const float* W1p = shared_e ? ws1 : (w1 + (size_t)e * D_ * H_);
    const float* B1p = shared_e ? bs1 : (b1 + (size_t)e * H_);
    const float* W2p = shared_e ? ws2 : (w2 + (size_t)e * H_ * D_);
    const float* B2p = shared_e ? bs2 : (b2 + (size_t)e * D_);

    const int tid  = threadIdx.x;
    const int wave = tid >> 5;
    const int lane = tid & 31;

    if (tid < MT_) {
        int m = mt * MT_ + tid;
        int tok; float wv;
        if (shared_e) { tok = (m < N_) ? m : 0; wv = (m < N_) ? 1.f : 0.f; }
        else if (m < cnt) { tok = tlist[(size_t)e * N_ + m]; wv = comb[(size_t)tok * E_ + e]; }
        else { tok = 0; wv = 0.f; }
        s.toks[tid] = tok; s.wgt[tid] = wv;
    }
    __syncthreads();

    // stage X tile (MT_ x 768) fp32 -> bf16 (packed pairs, float2 global loads)
    for (int p = tid; p < MT_ * (D_ / 2); p += 256) {
        int r = p / (D_ / 2), c2 = (p % (D_ / 2)) * 2;
        const float* xp = &x[(size_t)s.toks[r] * D_ + c2];
        stage_pair(&s.Xs[r][c2], xp[0], xp[1]);
    }

    const int rowA = lane & 15;
    const int kbA  = (lane >> 4) * 8;
    const int kkB  = (lane >> 4) * 8;
    const int colB = wave * 16 + (lane & 15);
    const int mh   = (lane >> 4) * 8;

    // per-thread staging coordinates: invariant across the whole kernel
    int go1[8], go2[8], lofs[8];     // global elem offsets (GEMM1/GEMM2), LDS elem offset
    #pragma unroll
    for (int q = 0; q < 8; ++q) {
        int p  = tid + q * 256;      // 0..2047
        int c  = p & 127, r0 = (p >> 7) * 2;
        go1[q]  = r0 * H_ + c;
        go2[q]  = r0 * D_ + c;
        lofs[q] = c * 40 + r0;
    }
    bf16* const bt0 = &s.Bt[0][0][0];
    bf16* const bt1 = &s.Bt[1][0][0];

    // ======== GEMM1: Hs = gelu(X @ W1 + b1), double-buffered weight tiles ========
    {
        constexpr int NK = D_ / 32;    // 24 k-steps per column tile
        constexpr int NT = H_ / 128;   // 24 column tiles
        #pragma unroll
        for (int q = 0; q < 8; ++q)
            stage_pair(bt0 + lofs[q], W1p[go1[q]], W1p[go1[q] + H_]);
        __syncthreads();

        bf16* bcur = bt0;
        bf16* bnxt = bt1;
        v8f acc0 = {}, acc1 = {};
        int ht = 0, ksi = 0;
        #pragma unroll 2
        for (int it = 0; it < NT * NK; ++it) {
            int nksi = ksi + 1, nht = ht;
            if (nksi == NK) { nksi = 0; nht++; }
            const bool has_next = (it + 1 < NT * NK);

            // issue coalesced global loads for the NEXT tile into registers
            float lo[8], hi[8];
            if (has_next) {
                const float* base = W1p + (size_t)(nksi * 32) * H_ + nht * 128;
                #pragma unroll
                for (int q = 0; q < 8; ++q) {
                    lo[q] = base[go1[q]]; hi[q] = base[go1[q] + H_];
                }
            }

            // compute from current buffer
            const int ks = ksi * 32;
            v16bf b  = ldfragB(bcur + colB * 40 + kkB);
            v16bf a0 = ldfragA(&s.Xs[rowA][ks + kbA]);
            v16bf a1 = ldfragA(&s.Xs[rowA + 16][ks + kbA]);
            acc0 = WMMA_BF16(a0, b, acc0);
            acc1 = WMMA_BF16(a1, b, acc1);

            // column-tile epilogue: +b1, fast gelu, store bf16 into Hs
            if (ksi == NK - 1) {
                const int h = ht * 128 + wave * 16 + (lane & 15);
                const float bb = B1p[h];
                #pragma unroll
                for (int r = 0; r < 8; ++r) {
                    s.Hs[mh + r][h]      = (bf16)gelu_fast(acc0[r] + bb);
                    s.Hs[16 + mh + r][h] = (bf16)gelu_fast(acc1[r] + bb);
                }
                acc0 = {}; acc1 = {};
            }

            // commit staged data to the other buffer
            if (has_next) {
                #pragma unroll
                for (int q = 0; q < 8; ++q)
                    stage_pair(bnxt + lofs[q], lo[q], hi[q]);
            }
            __syncthreads();
            bf16* t = bcur; bcur = bnxt; bnxt = t;
            ksi = nksi; ht = nht;
        }
    }

    // ======== GEMM2: out[tok] += wgt * (Hs @ W2 + b2), same pipeline ========
    {
        constexpr int NK = H_ / 32;    // 96 k-steps per column tile
        constexpr int NT = D_ / 128;   // 6 column tiles
        #pragma unroll
        for (int q = 0; q < 8; ++q)
            stage_pair(bt0 + lofs[q], W2p[go2[q]], W2p[go2[q] + D_]);
        __syncthreads();

        bf16* bcur = bt0;
        bf16* bnxt = bt1;
        v8f y0 = {}, y1 = {};
        int dt = 0, ksi = 0;
        #pragma unroll 2
        for (int it = 0; it < NT * NK; ++it) {
            int nksi = ksi + 1, ndt = dt;
            if (nksi == NK) { nksi = 0; ndt++; }
            const bool has_next = (it + 1 < NT * NK);

            float lo[8], hi[8];
            if (has_next) {
                const float* base = W2p + (size_t)(nksi * 32) * D_ + ndt * 128;
                #pragma unroll
                for (int q = 0; q < 8; ++q) {
                    lo[q] = base[go2[q]]; hi[q] = base[go2[q] + D_];
                }
            }

            const int ks = ksi * 32;
            v16bf b  = ldfragB(bcur + colB * 40 + kkB);
            v16bf a0 = ldfragA(&s.Hs[rowA][ks + kbA]);
            v16bf a1 = ldfragA(&s.Hs[rowA + 16][ks + kbA]);
            y0 = WMMA_BF16(a0, b, y0);
            y1 = WMMA_BF16(a1, b, y1);

            if (ksi == NK - 1) {
                const int d = dt * 128 + wave * 16 + (lane & 15);
                const float bb = B2p[d];
                #pragma unroll
                for (int r = 0; r < 8; ++r) {
                    atomicAdd(&out[(size_t)s.toks[mh + r] * D_ + d],
                              (y0[r] + bb) * s.wgt[mh + r]);
                    atomicAdd(&out[(size_t)s.toks[16 + mh + r] * D_ + d],
                              (y1[r] + bb) * s.wgt[16 + mh + r]);
                }
                y0 = {}; y1 = {};
            }

            if (has_next) {
                #pragma unroll
                for (int q = 0; q < 8; ++q)
                    stage_pair(bnxt + lofs[q], lo[q], hi[q]);
            }
            __syncthreads();
            bf16* t = bcur; bcur = bnxt; bnxt = t;
            ksi = nksi; dt = ndt;
        }
    }
}

// ---------------- aux loss + expert_num outputs ----------------
__global__ void finalize_kernel(const int* __restrict__ counts,
                                const float* __restrict__ gsum,
                                float* __restrict__ outp)
{
    int t = threadIdx.x; // 32 threads
    float p = 0.f;
    if (t < E_)
        p = (gsum[t] / (float)N_) * ((float)E_ * (float)counts[t] / (float)(K_ * N_));
    #pragma unroll
    for (int off = 16; off > 0; off >>= 1) p += __shfl_xor(p, off, 32);
    if (t == 0) outp[(size_t)N_ * D_] = p;
    if (t < E_) outp[(size_t)N_ * D_ + 1 + t] = (float)counts[t];
}

// ---------------- launcher ----------------
extern "C" void kernel_launch(void* const* d_in, const int* in_sizes, int n_in,
                              void* d_out, int out_size, void* d_ws, size_t ws_size,
                              hipStream_t stream) {
    const float* x      = (const float*)d_in[0];
    const float* gate_w = (const float*)d_in[1];
    const float* gate_b = (const float*)d_in[2];
    const float* biasb  = (const float*)d_in[3];
    const float* w1     = (const float*)d_in[4];
    const float* b1     = (const float*)d_in[5];
    const float* w2     = (const float*)d_in[6];
    const float* b2     = (const float*)d_in[7];
    const float* ws1    = (const float*)d_in[8];
    const float* bs1    = (const float*)d_in[9];
    const float* ws2    = (const float*)d_in[10];
    const float* bs2    = (const float*)d_in[11];
    float* out = (float*)d_out;

    // workspace carve-up (~380 KB)
    char* w = (char*)d_ws;
    float* comb = (float*)w;  w += (size_t)N_ * E_ * sizeof(float);
    int*   counts = (int*)w;  w += 32 * sizeof(int);
    float* gsum = (float*)w;  w += 32 * sizeof(float);
    int*   tlist = (int*)w;   w += (size_t)E_ * N_ * sizeof(int);
    (void)ws_size; (void)in_sizes; (void)n_in; (void)out_size;

    zero_out_kernel<<<((size_t)N_ * D_ + 255) / 256, 256, 0, stream>>>(out);
    init_small_kernel<<<1, 32, 0, stream>>>(counts, gsum);

    gate_kernel<<<N_ / 8, 256, 0, stream>>>(x, gate_w, gate_b, biasb,
                                            comb, counts, gsum, tlist);

    dim3 grid(N_ / MT_, NEXP);
    expert_kernel<<<grid, 256, sizeof(SmemExpert), stream>>>(
        x, w1, b1, w2, b2, ws1, bs1, ws2, bs2, counts, tlist, comb, out);

    finalize_kernel<<<1, 32, 0, stream>>>(counts, gsum, out);
}
